// Inter_sub_sim_v1_48756468744575
// MI455X (gfx1250) — matrix operand order repaired and checked by reading, not compiled, not run on hardware
//
#include <hip/hip_runtime.h>
#include <hip/hip_bf16.h>
#include <math.h>

#define BDIM 32768
#define DDIM 512
#define KSUB 256
#define TWOB (2 * BDIM)

typedef __attribute__((ext_vector_type(16))) __bf16 v16bf;
typedef __attribute__((ext_vector_type(8)))  __bf16 v8bf;
typedef __attribute__((ext_vector_type(8)))  float  v8f;
typedef __attribute__((ext_vector_type(4)))  float  f4v;

// cf = concat(X[:,0,:], X[:,1,:]) -> row i of cf inside X [B,2,D]
__device__ __forceinline__ const float* cf_row(const float* __restrict__ X, int i) {
  return X + ((i < BDIM) ? ((size_t)i * (2 * DDIM))
                         : ((size_t)(i - BDIM) * (2 * DDIM) + DDIM));
}
__device__ __forceinline__ int seg_of(const int* __restrict__ sub, int i) {
  return sub[(i < BDIM) ? i : (i - BDIM)];
}

// CDNA5 async global->LDS DMA: copy 64B (4 x b128) per thread, tracked by ASYNCcnt.
__device__ __forceinline__ void stage64B_async(unsigned lds_off, const void* gaddr) {
  unsigned long long g = (unsigned long long)gaddr;
  asm volatile(
      "global_load_async_to_lds_b128 %0, %1, off\n\t"
      "global_load_async_to_lds_b128 %0, %1, off offset:16\n\t"
      "global_load_async_to_lds_b128 %0, %1, off offset:32\n\t"
      "global_load_async_to_lds_b128 %0, %1, off offset:48"
      :: "v"(lds_off), "v"(g)
      : "memory");
}
__device__ __forceinline__ void wait_async0() {
  asm volatile("s_wait_asynccnt 0x0" ::: "memory");
}

// ---------------- init scratch ----------------
__global__ void k_init(int* cnt, int* cursor, float* sqsum, float* scal) {
  int t = threadIdx.x;
  cnt[t] = 0; cursor[t] = 0; sqsum[t] = 0.0f;
  if (t < 3) scal[t] = 0.0f;   // [0]=||cf||^2  [1]=||cen||^2  [2]=loss sum
}

// ---------------- per-subject histogram ----------------
__global__ void k_count(const int* __restrict__ sub, int* __restrict__ cnt) {
  int i = blockIdx.x * blockDim.x + threadIdx.x;
  int stride = gridDim.x * blockDim.x;
  for (; i < TWOB; i += stride) atomicAdd(&cnt[seg_of(sub, i)], 1);
}

// exclusive prefix over 256 counts
__global__ void k_scan(const int* __restrict__ cnt, int* __restrict__ offs) {
  __shared__ int s[KSUB];
  int t = threadIdx.x;
  s[t] = cnt[t];
  __syncthreads();
  for (int d = 1; d < KSUB; d <<= 1) {
    int u = (t >= d) ? s[t - d] : 0;
    __syncthreads();
    s[t] += u;
    __syncthreads();
  }
  offs[t] = s[t] - cnt[t];
}

__global__ void k_scatter(const int* __restrict__ sub, const int* __restrict__ offs,
                          int* __restrict__ cursor, int* __restrict__ rowlist) {
  int i = blockIdx.x * blockDim.x + threadIdx.x;
  int stride = gridDim.x * blockDim.x;
  for (; i < TWOB; i += stride) {
    int k = seg_of(sub, i);
    int pos = atomicAdd(&cursor[k], 1);
    rowlist[offs[k] + pos] = i;
  }
}

// ---------------- centroids (block per subject) ----------------
__global__ void k_centroid(const float* __restrict__ X, const int* __restrict__ rowlist,
                           const int* __restrict__ offs, const int* __restrict__ cnt,
                           float* __restrict__ cen, __bf16* __restrict__ cenb,
                           float* __restrict__ censq) {
  __shared__ float red[256];
  int k = blockIdx.x, t = threadIdx.x;
  int off = offs[k], c = cnt[k];
  float a0 = 0.0f, a1 = 0.0f;
  for (int j = 0; j < c; ++j) {
    const float* rp = cf_row(X, rowlist[off + j]);
    a0 += rp[t];
    a1 += rp[t + 256];
  }
  float inv = (c > 0) ? 1.0f / (float)c : 0.0f;
  a0 *= inv; a1 *= inv;
  size_t base = (size_t)k * DDIM;
  cen[base + t] = a0;            cen[base + t + 256] = a1;
  cenb[base + t] = (__bf16)a0;   cenb[base + t + 256] = (__bf16)a1;
  red[t] = a0 * a0 + a1 * a1;
  __syncthreads();
  for (int s = 128; s > 0; s >>= 1) {
    if (t < s) red[t] += red[t + s];
    __syncthreads();
  }
  if (t == 0) atomicAdd(censq, red[0]);
}

// ---------------- distance pass (wave per row), fuses ||cf||^2 ----------------
__global__ void k_dist(const float* __restrict__ X, const int* __restrict__ sub,
                       const float* __restrict__ cen, float* __restrict__ sqsum,
                       float* __restrict__ cfsq) {
  int wave = threadIdx.x >> 5, lane = threadIdx.x & 31;
  int row = blockIdx.x * 8 + wave;
  int s = seg_of(sub, row);
  const float* rp = cf_row(X, row);
  const float* cp = cen + (size_t)s * DDIM;
  float acc = 0.0f, nrm = 0.0f;
  for (int d = lane; d < DDIM; d += 32) {
    float x = rp[d];
    float df = x - cp[d];
    acc += df * df;
    nrm += x * x;
  }
#pragma unroll
  for (int m = 16; m > 0; m >>= 1) {
    acc += __shfl_xor(acc, m, 32);
    nrm += __shfl_xor(nrm, m, 32);
  }
  if (lane == 0) {
    atomicAdd(&sqsum[s], sqrtf(sqrtf(acc)));   // sqrt(dist) = (||.||^2)^(1/4)
    atomicAdd(cfsq, nrm);
  }
}

// ---------------- density: quantile clip, write invd[col] ----------------
__global__ void k_density(const int* __restrict__ cnt, const float* __restrict__ sqsum,
                          const float* __restrict__ scal, float* __restrict__ invd) {
  __shared__ float red[KSUB];
  __shared__ float srt[KSUB];
  int t = threadIdx.x;
  float c = (float)cnt[t];
  bool valid = c > 1.0f;
  float dens = valid ? (sqsum[t] / c) / logf(c + 10.0f) : -__builtin_inff();
  red[t] = dens;
  __syncthreads();
  for (int s = 128; s > 0; s >>= 1) {
    if (t < s) red[t] = fmaxf(red[t], red[t + s]);
    __syncthreads();
  }
  float dmax = red[0];
  __syncthreads();
  if (!valid) dens = dmax;
  srt[t] = dens;
  __syncthreads();
  // bitonic sort ascending (256 elements, 256 threads)
  for (int ks = 2; ks <= KSUB; ks <<= 1) {
    for (int j = ks >> 1; j > 0; j >>= 1) {
      int ixj = t ^ j;
      if (ixj > t) {
        float a = srt[t], b = srt[ixj];
        bool up = ((t & ks) == 0);
        if ((a > b) == up) { srt[t] = b; srt[ixj] = a; }
      }
      __syncthreads();
    }
  }
  // linear-interpolation quantiles at q=0.1, 0.9 (jnp.quantile default)
  float p10 = 0.1f * (KSUB - 1); int l10 = (int)p10; float f10 = p10 - (float)l10;
  float q10 = srt[l10] + f10 * (srt[l10 + 1] - srt[l10]);
  float p90 = 0.9f * (KSUB - 1); int l90 = (int)p90; float f90 = p90 - (float)l90;
  float q90 = srt[l90] + f90 * (srt[l90 + 1] - srt[l90]);
  float cl = fminf(fmaxf(dens, q10), q90);
  red[t] = cl;
  __syncthreads();
  for (int s = 128; s > 0; s >>= 1) {
    if (t < s) red[t] += red[t + s];
    __syncthreads();
  }
  float mean = red[0] * (1.0f / KSUB);
  float densf = 0.1f * cl / mean;
  float invG = rsqrtf(scal[0]) * rsqrtf(scal[1]);  // 1/(||cf||_F * ||cen||_F)
  invd[t] = invG / densf;
}

// ---------------- fused GEMM (bf16 WMMA) + masked log-softmax + loss ----------------
// Block: 256 threads = 8 waves; each wave: 16 rows x all 256 columns.
// Double-buffered LDS stages [256 cols][32 k] bf16 slices of cen via async
// global->LDS DMA; B-fragment ds_loads are software-pipelined one tile ahead.
__global__ void __launch_bounds__(256, 2)
k_gemm_loss(const float* __restrict__ X, const int* __restrict__ sub,
            const __bf16* __restrict__ cenb, const float* __restrict__ invd,
            float* __restrict__ loss_sum) {
  __shared__ __align__(64) __bf16 sB[2][KSUB * 40];  // 2 x 20KB, 80B per column
  const int tid  = threadIdx.x;
  const int wave = tid >> 5;
  const int lane = tid & 31;
  const int n    = lane & 15;
  const int hi   = lane >> 4;
  const int rowBase = (blockIdx.x * 8 + wave) * 16;
  const float* arow = cf_row(X, rowBase + n);  // A lane holds row m = lane&15

  v8f acc[16];
#pragma unroll
  for (int t = 0; t < 16; ++t)
#pragma unroll
    for (int r = 0; r < 8; ++r) acc[t][r] = 0.0f;

  // prologue: async-stage k-slice 0 into buffer 0 (thread tid owns column tid)
  const __bf16* gcol = cenb + (size_t)tid * DDIM;
  stage64B_async((unsigned)(size_t)(&sB[0][tid * 40]), gcol);

  for (int kb = 0; kb < DDIM; kb += 32) {
    wait_async0();        // own async writes to LDS landed
    __syncthreads();      // everyone's writes visible; prev buffer reads done
    const int cur = (kb >> 5) & 1;
    if (kb + 32 < DDIM)   // overlap next slice DMA with this slice's WMMAs
      stage64B_async((unsigned)(size_t)(&sB[cur ^ 1][tid * 40]), gcol + kb + 32);

    // A fragment: lane m, elements j: K = kb + hi*16 + j
    f4v a0 = *(const f4v*)(arow + kb + hi * 16);
    f4v a1 = *(const f4v*)(arow + kb + hi * 16 + 4);
    f4v a2 = *(const f4v*)(arow + kb + hi * 16 + 8);
    f4v a3 = *(const f4v*)(arow + kb + hi * 16 + 12);
    v16bf af;
    af[0]  = (__bf16)a0[0]; af[1]  = (__bf16)a0[1]; af[2]  = (__bf16)a0[2]; af[3]  = (__bf16)a0[3];
    af[4]  = (__bf16)a1[0]; af[5]  = (__bf16)a1[1]; af[6]  = (__bf16)a1[2]; af[7]  = (__bf16)a1[3];
    af[8]  = (__bf16)a2[0]; af[9]  = (__bf16)a2[1]; af[10] = (__bf16)a2[2]; af[11] = (__bf16)a2[3];
    af[12] = (__bf16)a3[0]; af[13] = (__bf16)a3[1]; af[14] = (__bf16)a3[2]; af[15] = (__bf16)a3[3];

    // B fragments: lane holds col = t*16+n, K = kb + hi*16 + j (contiguous 16).
    // Software pipeline: prefetch tile t+1's pair before tile t's WMMA.
    const __bf16* bbase = &sB[cur][n * 40 + hi * 16];
    v8bf b0 = *(const v8bf*)(bbase);
    v8bf b1 = *(const v8bf*)(bbase + 8);
#pragma unroll
    for (int t = 0; t < 16; ++t) {
      v8bf nb0, nb1;
      if (t < 15) {
        const __bf16* bp = bbase + (t + 1) * (16 * 40);
        nb0 = *(const v8bf*)(bp);
        nb1 = *(const v8bf*)(bp + 8);
      }
      v16bf bf;
#pragma unroll
      for (int j = 0; j < 8; ++j) { bf[j] = b0[j]; bf[8 + j] = b1[j]; }
      acc[t] = __builtin_amdgcn_wmma_f32_16x16x32_bf16(
          false, af, false, bf, (short)0, acc[t], false, false);
      if (t < 15) { b0 = nb0; b1 = nb1; }
    }
  }

  // ---- masked log-softmax over 256 columns, done in the C-register layout ----
  float invd_r[16];
#pragma unroll
  for (int t = 0; t < 16; ++t) invd_r[t] = invd[t * 16 + n];
  int segr[8];
#pragma unroll
  for (int r = 0; r < 8; ++r) segr[r] = seg_of(sub, rowBase + r + 8 * hi);

  float lossA = 0.0f;
#pragma unroll
  for (int r = 0; r < 8; ++r) {
    float vals[16];
    float rmax = -__builtin_inff();
    float s2 = 0.0f;
#pragma unroll
    for (int t = 0; t < 16; ++t) {
      int col = t * 16 + n;
      float v = acc[t][r] * invd_r[t];
      bool msk = (col == segr[r]);       // row's own subject column is excluded
      vals[t] = msk ? -__builtin_inff() : v;
      if (!msk) { rmax = fmaxf(rmax, v); s2 += v; }
    }
#pragma unroll
    for (int m = 1; m < 16; m <<= 1) rmax = fmaxf(rmax, __shfl_xor(rmax, m, 32));
    float s1 = 0.0f;
#pragma unroll
    for (int t = 0; t < 16; ++t)
      if (vals[t] > -__builtin_inff()) s1 += __expf(vals[t] - rmax);
#pragma unroll
    for (int m = 1; m < 16; m <<= 1) {
      s1 += __shfl_xor(s1, m, 32);
      s2 += __shfl_xor(s2, m, 32);
    }
    // loss_row = lse - mean(sim over kept) ; lse = rmax + log(sum exp)
    lossA += rmax + __logf(s1) - s2 * (1.0f / (KSUB - 1));
  }
  if (n == 0) atomicAdd(loss_sum, lossA);
}

__global__ void k_final(const float* __restrict__ loss_sum, float* __restrict__ out) {
  out[0] = loss_sum[0] / (float)TWOB;
}

extern "C" void kernel_launch(void* const* d_in, const int* in_sizes, int n_in,
                              void* d_out, int out_size, void* d_ws, size_t ws_size,
                              hipStream_t stream) {
  const float* X  = (const float*)d_in[0];   // [B, 2, D] f32
  const int* sub  = (const int*)d_in[1];     // [B] int32
  float* out      = (float*)d_out;           // scalar f32

  char* p = (char*)d_ws;
  float*  cen     = (float*)(p);                 // 256*512*4  = 524288 B
  __bf16* cenb    = (__bf16*)(p + 524288);       // 256*512*2  = 262144 B
  int*    rowlist = (int*)(p + 786432);          // 65536*4    = 262144 B
  int*    cnt     = (int*)(p + 1048576);         // 1024 B
  int*    offs    = (int*)(p + 1049600);         // 1024 B
  int*    cursor  = (int*)(p + 1050624);         // 1024 B
  float*  sqsum   = (float*)(p + 1051648);       // 1024 B
  float*  invd    = (float*)(p + 1052672);       // 1024 B
  float*  scal    = (float*)(p + 1053696);       // 3 floats

  k_init<<<1, 256, 0, stream>>>(cnt, cursor, sqsum, scal);
  k_count<<<256, 256, 0, stream>>>(sub, cnt);
  k_scan<<<1, 256, 0, stream>>>(cnt, offs);
  k_scatter<<<256, 256, 0, stream>>>(sub, offs, cursor, rowlist);
  k_centroid<<<KSUB, 256, 0, stream>>>(X, rowlist, offs, cnt, cen, cenb, &scal[1]);
  k_dist<<<TWOB / 8, 256, 0, stream>>>(X, sub, cen, sqsum, &scal[0]);
  k_density<<<1, 256, 0, stream>>>(cnt, sqsum, scal, invd);
  k_gemm_loss<<<TWOB / 128, 256, 0, stream>>>(X, sub, cenb, invd, &scal[2]);
  k_final<<<1, 1, 0, stream>>>(&scal[2], out);
}
